// Seq2Seq_Model_3307124818138
// MI455X (gfx1250) — compile-verified, matrix operand order checked by reading, not run on hardware
//
#include <hip/hip_runtime.h>
#include <hip/hip_bf16.h>
#include <math.h>

// ---------------------------------------------------------------------------
// Seq2Seq LSTM (encoder/decoder) + vocab projection for MI455X (gfx1250).
//  - input projections hoisted into batched f32-WMMA GEMMs
//  - recurrence: ONE fused kernel/step: 4-gate WMMA GEMM + nonlinearity,
//    h ping-pongs between two buffers (c is elementwise-owned, in place)
//  - final FC: bf16x3 split GEMM on v_wmma_f32_16x16x32_bf16, 64x16
//    macro-tiles so the fp32->bf16 hi/lo weight split is amortized 4x
// ---------------------------------------------------------------------------

#define EMB 256
#define HID 512
#define G4H 2048            // 4*HID
#define BSZ 32
#define LSEQ 64
#define VY  32000
#define FCMT 4              // M-tiles per wave in the FC GEMM

typedef __attribute__((ext_vector_type(2)))  float  v2f;
typedef __attribute__((ext_vector_type(8)))  float  v8f;
typedef __attribute__((ext_vector_type(8)))  __bf16 v8bf;
typedef __attribute__((ext_vector_type(16))) __bf16 v16bf;

static __device__ __forceinline__ v8f wmma_f32(v2f a, v2f b, v8f c) {
  return __builtin_amdgcn_wmma_f32_16x16x4_f32(false, a, false, b, (short)0, c,
                                               false, false);
}
static __device__ __forceinline__ v8f wmma_bf16(v16bf a, v16bf b, v8f c) {
  return __builtin_amdgcn_wmma_f32_16x16x32_bf16(false, a, false, b, (short)0, c,
                                                 false, false);
}

// ---------------------------------------------------------------------------
// Embedding gather.  row = t*B + b.  Decoder quirk: token index = max(t-1,0).
// ---------------------------------------------------------------------------
__global__ void gather_embed(const int* __restrict__ tok,     // [B, L]
                             const float* __restrict__ emb,   // [V, EMB]
                             float* __restrict__ out,         // [L*B, EMB]
                             int dec_quirk) {
  int row = blockIdx.x;
  int t = row / BSZ, b = row % BSZ;
  int idx = dec_quirk ? (t == 0 ? 0 : t - 1) : t;
  int token = tok[b * LSEQ + idx];
  const float4* src = reinterpret_cast<const float4*>(emb + (size_t)token * EMB);
  float4*       dst = reinterpret_cast<float4*>(out + (size_t)row * EMB);
  dst[threadIdx.x] = src[threadIdx.x];
}

// ---------------------------------------------------------------------------
// Batched input projection: C[m][n] = sum_k A[m][k] * W[n][k] + bias[n]
// A:[M,K=EMB], W:[4H,K], C:[M,4H].  One wave per 16x16 tile, exact fp32 WMMA.
// ---------------------------------------------------------------------------
__global__ __launch_bounds__(128)
void xproj_gemm(const float* __restrict__ A, const float* __restrict__ W,
                const float* __restrict__ bias, float* __restrict__ C,
                int M, int N, int K) {
  int wave = blockIdx.x * (blockDim.x >> 5) + (threadIdx.x >> 5);
  int lane = threadIdx.x & 31;
  int ntiles = N >> 4;
  int tm = wave / ntiles, tn = wave % ntiles;
  int half = lane >> 4;      // 0: K base; 1: K base+2 (A/B frag upper half)
  int r    = lane & 15;
  const float* arow = A + (size_t)(tm * 16 + r) * K;
  const float* brow = W + (size_t)(tn * 16 + r) * K;
  v8f c = {};
  for (int k0 = 0; k0 < K; k0 += 4) {
    int kk = k0 + half * 2;
    c = wmma_f32(*reinterpret_cast<const v2f*>(arow + kk),
                 *reinterpret_cast<const v2f*>(brow + kk), c);
  }
  int col = tn * 16 + r;
  float bv = bias[col];
  #pragma unroll
  for (int e = 0; e < 8; ++e) {
    int row = tm * 16 + half * 8 + e;          // C frag: VGPR e -> M = e (+8 hi half)
    C[(size_t)row * N + col] = c[e] + bv;
  }
}

// ---------------------------------------------------------------------------
// Fused LSTM step.  Each wave owns a 16-wide hidden slice (tn) and a 16-row
// batch slice (tm): accumulates the i/f/g/o 16x16 tiles with 4 WMMA
// accumulators sharing one A-fragment load, then applies the gate
// nonlinearities and writes c (in place) and h (ping-pong).  Decoder steps
// also emit h as bf16 hi/lo at row m = b*Ly + l for the FC GEMM.
// 64 waves total (tm in {0,1}, tn in 0..31).
// ---------------------------------------------------------------------------
__global__ __launch_bounds__(128)
void step_fused(const float* __restrict__ h_in,     // [32, HID]
                const float* __restrict__ Whh,      // [4H, HID]
                const float* __restrict__ xproj_t,  // [32, 4H] (includes bias)
                float* __restrict__ cbuf,           // [32, HID] in/out
                float* __restrict__ h_out,          // [32, HID]
                __bf16* __restrict__ hsHi, __bf16* __restrict__ hsLo,
                int l) {
  int wave = blockIdx.x * (blockDim.x >> 5) + (threadIdx.x >> 5);
  int lane = threadIdx.x & 31;
  const int ntiles = HID >> 4;                     // 32
  int tm = wave / ntiles, tn = wave % ntiles;
  int half = lane >> 4, r = lane & 15;
  const float* arow = h_in + (size_t)(tm * 16 + r) * HID;
  const float* bi = Whh + (size_t)(0 * HID + tn * 16 + r) * HID;
  const float* bf = Whh + (size_t)(1 * HID + tn * 16 + r) * HID;
  const float* bg = Whh + (size_t)(2 * HID + tn * 16 + r) * HID;
  const float* bo = Whh + (size_t)(3 * HID + tn * 16 + r) * HID;
  v8f ai = {}, af = {}, ag = {}, ao = {};
  for (int k0 = 0; k0 < HID; k0 += 4) {
    int kk = k0 + half * 2;
    v2f a = *reinterpret_cast<const v2f*>(arow + kk);
    ai = wmma_f32(a, *reinterpret_cast<const v2f*>(bi + kk), ai);
    af = wmma_f32(a, *reinterpret_cast<const v2f*>(bf + kk), af);
    ag = wmma_f32(a, *reinterpret_cast<const v2f*>(bg + kk), ag);
    ao = wmma_f32(a, *reinterpret_cast<const v2f*>(bo + kk), ao);
  }
  int col = tn * 16 + r;                           // hidden index k
  #pragma unroll
  for (int e = 0; e < 8; ++e) {
    int b = tm * 16 + half * 8 + e;                // batch row
    const float* xp = xproj_t + (size_t)b * G4H;
    float gi = ai[e] + xp[col];
    float gf = af[e] + xp[HID + col];
    float gg = ag[e] + xp[2 * HID + col];
    float go = ao[e] + xp[3 * HID + col];
    float si = 1.f / (1.f + __expf(-gi));
    float sf = 1.f / (1.f + __expf(-gf));
    float so = 1.f / (1.f + __expf(-go));
    size_t hc = (size_t)b * HID + col;
    float cn = sf * cbuf[hc] + si * tanhf(gg);
    float hn = so * tanhf(cn);
    cbuf[hc] = cn;
    h_out[hc] = hn;
    if (hsHi) {
      size_t m = (size_t)b * LSEQ + l;
      __bf16 hi = (__bf16)hn;
      hsHi[m * HID + col] = hi;
      hsLo[m * HID + col] = (__bf16)(hn - (float)hi);
    }
  }
}

// ---------------------------------------------------------------------------
// Final FC: logits[m][v] = sum_k hs[m][k] * fc_W[v][k] + fc_b[v]
// M=2048, N=32000, K=512.  bf16x3 split: C += Ahi*Bhi + Ahi*Blo + Alo*Bhi.
// Each wave computes a 64x16 macro-tile (FCMT=4 M-tiles) so the in-register
// fp32->bf16 hi/lo weight split is amortized across 12 WMMAs per K-chunk.
// ---------------------------------------------------------------------------
__global__ __launch_bounds__(128)
void fc_gemm(const __bf16* __restrict__ Ahi, const __bf16* __restrict__ Alo, // [2048, HID]
             const float* __restrict__ W,                                    // [VY, HID]
             const float* __restrict__ bias,                                 // [VY]
             float* __restrict__ out) {                                      // [2048, VY]
  int wave = blockIdx.x * (blockDim.x >> 5) + (threadIdx.x >> 5);
  int lane = threadIdx.x & 31;
  const int ntiles = VY >> 4;                      // 2000
  int tmg = wave / ntiles, tn = wave % ntiles;     // tmg: 0..31 (64 rows each)
  int half = lane >> 4, r = lane & 15;
  const __bf16* ah[FCMT];
  const __bf16* al[FCMT];
  #pragma unroll
  for (int mt = 0; mt < FCMT; ++mt) {
    size_t row = (size_t)(tmg * (16 * FCMT) + mt * 16 + r);
    ah[mt] = Ahi + row * HID;
    al[mt] = Alo + row * HID;
  }
  const float* wr = W + (size_t)(tn * 16 + r) * HID;
  v8f acc[FCMT] = {};
  for (int k0 = 0; k0 < HID; k0 += 32) {
    int kb = k0 + half * 8;                        // lanes>=16: K base+8 / base+24
    // ---- B fragment from fp32 weights, hi/lo split once per chunk --------
    float wv[16];
    *reinterpret_cast<float4*>(&wv[0])  = *reinterpret_cast<const float4*>(wr + kb);
    *reinterpret_cast<float4*>(&wv[4])  = *reinterpret_cast<const float4*>(wr + kb + 4);
    *reinterpret_cast<float4*>(&wv[8])  = *reinterpret_cast<const float4*>(wr + kb + 16);
    *reinterpret_cast<float4*>(&wv[12]) = *reinterpret_cast<const float4*>(wr + kb + 20);
    v16bf bH, bL;
    #pragma unroll
    for (int j = 0; j < 16; ++j) {
      __bf16 hb = (__bf16)wv[j];
      bH[j] = hb;
      bL[j] = (__bf16)(wv[j] - (float)hb);
    }
    // ---- 4 A macro-rows reuse the converted B fragments ------------------
    #pragma unroll
    for (int mt = 0; mt < FCMT; ++mt) {
      v8bf a0 = *reinterpret_cast<const v8bf*>(ah[mt] + kb);
      v8bf a1 = *reinterpret_cast<const v8bf*>(ah[mt] + kb + 16);
      v8bf l0 = *reinterpret_cast<const v8bf*>(al[mt] + kb);
      v8bf l1 = *reinterpret_cast<const v8bf*>(al[mt] + kb + 16);
      v16bf aH, aL;
      #pragma unroll
      for (int j = 0; j < 8; ++j) {
        aH[j] = a0[j]; aH[j + 8] = a1[j];
        aL[j] = l0[j]; aL[j + 8] = l1[j];
      }
      acc[mt] = wmma_bf16(aH, bH, acc[mt]);
      acc[mt] = wmma_bf16(aH, bL, acc[mt]);
      acc[mt] = wmma_bf16(aL, bH, acc[mt]);
    }
  }
  int col = tn * 16 + r;
  float bv = bias[col];
  #pragma unroll
  for (int mt = 0; mt < FCMT; ++mt) {
    #pragma unroll
    for (int e = 0; e < 8; ++e) {
      int row = tmg * (16 * FCMT) + mt * 16 + half * 8 + e;
      out[(size_t)row * VY + col] = acc[mt][e] + bv;
    }
  }
}

// ---------------------------------------------------------------------------
// Host-side orchestration (graph-capture safe: only async ops on `stream`)
// ---------------------------------------------------------------------------
extern "C" void kernel_launch(void* const* d_in, const int* in_sizes, int n_in,
                              void* d_out, int out_size, void* d_ws, size_t ws_size,
                              hipStream_t stream) {
  (void)in_sizes; (void)n_in; (void)out_size; (void)ws_size;
  const int*   X       = (const int*)d_in[0];
  const int*   y       = (const int*)d_in[1];
  // d_in[2] = teacher_forcing (always 1 in reference; ignored)
  const float* emb_X   = (const float*)d_in[3];
  const float* enc_Wih = (const float*)d_in[4];
  const float* enc_Whh = (const float*)d_in[5];
  const float* enc_b   = (const float*)d_in[6];
  const float* emb_y   = (const float*)d_in[7];
  const float* dec_Wih = (const float*)d_in[8];
  const float* dec_Whh = (const float*)d_in[9];
  const float* dec_b   = (const float*)d_in[10];
  const float* fc_W    = (const float*)d_in[11];
  const float* fc_b    = (const float*)d_in[12];
  float* out = (float*)d_out;

  // ---- workspace carve-up (all sizes multiples of 256B) ------------------
  char* ws = (char*)d_ws;
  float* Xe     = (float*)ws;   ws += (size_t)LSEQ * BSZ * EMB * 4;   //  2 MB
  float* Xd     = (float*)ws;   ws += (size_t)LSEQ * BSZ * EMB * 4;   //  2 MB
  float* xprojE = (float*)ws;   ws += (size_t)LSEQ * BSZ * G4H * 4;   // 16 MB
  float* xprojD = (float*)ws;   ws += (size_t)LSEQ * BSZ * G4H * 4;   // 16 MB
  float* h0     = (float*)ws;   ws += (size_t)BSZ * HID * 4;          // ping
  float* h1     = (float*)ws;   ws += (size_t)BSZ * HID * 4;          // pong
  float* cbuf   = (float*)ws;   ws += (size_t)BSZ * HID * 4;
  __bf16* hsHi  = (__bf16*)ws;  ws += (size_t)BSZ * LSEQ * HID * 2;   //  2 MB
  __bf16* hsLo  = (__bf16*)ws;  /* total ~40.2 MB */

  // ---- 1) embedding gathers ----------------------------------------------
  gather_embed<<<LSEQ * BSZ, EMB / 4, 0, stream>>>(X, emb_X, Xe, 0);
  gather_embed<<<LSEQ * BSZ, EMB / 4, 0, stream>>>(y, emb_y, Xd, 1);

  // ---- 2) batched input projections (x @ Wih^T + b) ----------------------
  {
    int M = LSEQ * BSZ;                       // 2048
    int tiles = (M / 16) * (G4H / 16);        // 16384 waves
    xproj_gemm<<<tiles / 4, 128, 0, stream>>>(Xe, enc_Wih, enc_b, xprojE, M, G4H, EMB);
    xproj_gemm<<<tiles / 4, 128, 0, stream>>>(Xd, dec_Wih, dec_b, xprojD, M, G4H, EMB);
  }

  // ---- 3) zero-init h (ping buffer) and c --------------------------------
  hipMemsetAsync(h0,   0, (size_t)BSZ * HID * 4, stream);
  hipMemsetAsync(cbuf, 0, (size_t)BSZ * HID * 4, stream);

  // ---- 4) recurrence: 64 encoder + 64 decoder fused steps ----------------
  const int step_waves = (BSZ / 16) * (HID / 16);  // 64 waves -> 16 blocks x 128
  float* hp[2] = {h0, h1};
  int p = 0;
  for (int t = 0; t < LSEQ; ++t, p ^= 1) {
    step_fused<<<step_waves / 4, 128, 0, stream>>>(
        hp[p], enc_Whh, xprojE + (size_t)t * BSZ * G4H, cbuf, hp[p ^ 1],
        nullptr, nullptr, 0);
  }
  for (int l = 0; l < LSEQ; ++l, p ^= 1) {
    step_fused<<<step_waves / 4, 128, 0, stream>>>(
        hp[p], dec_Whh, xprojD + (size_t)l * BSZ * G4H, cbuf, hp[p ^ 1],
        hsHi, hsLo, l);
  }

  // ---- 5) vocab projection: bf16x3 WMMA GEMM, 64x16 macro-tiles ----------
  {
    int waves = ((BSZ * LSEQ) / (16 * FCMT)) * (VY / 16);  // 32*2000 = 64000
    fc_gemm<<<waves / 4, 128, 0, stream>>>(hsHi, hsLo, fc_W, fc_b, out);
  }
}
